// ElmanRNN_42417097016404
// MI455X (gfx1250) — matrix operand order repaired and checked by reading, not compiled, not run on hardware
//
#include <hip/hip_runtime.h>

#define SEQ   2048
#define BATCH 64
#define IN_   256
#define HID   256

typedef __attribute__((ext_vector_type(16))) _Float16 v16h;
typedef __attribute__((ext_vector_type(8)))  _Float16 v8h;
typedef __attribute__((ext_vector_type(4)))  _Float16 v4h;
typedef __attribute__((ext_vector_type(8)))  float    v8f;

// ---------- fragment helpers (16x16x32 f16 WMMA operands; contiguous k order, same in A and B) ----------

// B fragment (K x N = 32 x 16): lane holds column n = lane%16.
// lanes 0-15: K = ks*32 + 0..15 ; lanes 16-31: K = ks*32 + 16..31 (contiguous per lane)
static __device__ __forceinline__ v16h load_b_f32(const float* __restrict__ wrow,
                                                  int ks, int hi16) {
  const float* w = wrow + ks * 32 + hi16 * 16;
  union { v16h v; _Float16 h[16]; } u;
#pragma unroll
  for (int i = 0; i < 16; ++i) u.h[i] = (_Float16)w[i];
  return u.v;
}

// A fragment (M x K = 16 x 32): lane holds row m = lane%16.
// elems 0-15: K = ks*32 + hi16*16 + 0..15 (contiguous per lane; two 16B ds_load_b128)
static __device__ __forceinline__ v16h load_a_lds(const _Float16* row, int ks, int hi16) {
  int kb = ks * 32 + hi16 * 16;
  v8h lo = *(const v8h*)(row + kb);
  v8h hh = *(const v8h*)(row + kb + 8);
  union { v16h v; _Float16 h[16]; } u;
#pragma unroll
  for (int i = 0; i < 8; ++i) { u.h[i] = lo[i]; u.h[i + 8] = hh[i]; }
  return u.v;
}

// tanh via v_exp_f32 + v_rcp_f32; exact saturation for large |x| (exp->inf => 1.0)
static __device__ __forceinline__ float fast_tanhf(float x) {
  float ax = __builtin_fabsf(x);
  float e  = __expf(2.0f * ax);
  float r  = 1.0f - 2.0f * __builtin_amdgcn_rcpf(e + 1.0f);
  return __builtin_copysignf(r, x);
}

// ====================== Kernel 1: xp = x @ W_ih^T + (b_ih + b_hh) ======================
// Grid 1024 x 512 threads (16 waves). Block stages 2 chunks of 64 rows of x as f16 in
// LDS; wave w owns n-tile w with W_ih B-fragments resident in VGPRs. Writes xp -> d_out.
#define K1_ROWS 64
#define K1_PAD  264   // padded f16 row stride (528B = 33*16B: aligned, bank-spread)

__global__ __launch_bounds__(512) void rnn_xproj_wmma(
    const float* __restrict__ x, const float* __restrict__ Wih,
    const float* __restrict__ bih, const float* __restrict__ bhh,
    float* __restrict__ xp)
{
  __shared__ __align__(16) _Float16 As[K1_ROWS * K1_PAD];
  const int tid  = threadIdx.x;
  const int lane = tid & 31;
  const int wv   = tid >> 5;       // n-tile 0..15
  const int hi16 = lane >> 4;
  const int la   = lane & 15;
  const int n    = wv * 16 + la;   // output column owned by this lane

  const float bsum = bih[n] + bhh[n];

  // W_ih fragments for all 8 k-slices, converted fp32->f16 once, kept in VGPRs
  const float* wrow = Wih + (size_t)n * IN_;
  v16h bf[8];
#pragma unroll
  for (int ks = 0; ks < 8; ++ks) bf[ks] = load_b_f32(wrow, ks, hi16);

#pragma unroll 1
  for (int j = 0; j < 2; ++j) {
    const int R = (blockIdx.x * 2 + j) * K1_ROWS;   // first global row of this chunk
    // Stage 64 x 256 fp32 -> f16 into LDS (512 threads * 8 float4 each)
    const float4* xs = (const float4*)(x + (size_t)R * IN_);
#pragma unroll
    for (int i = 0; i < 8; ++i) {
      int idx = tid + i * 512;          // 0..4095 float4s; row = idx/64, c4 = idx%64
      int row = idx >> 6;
      int c4  = idx & 63;
      float4 f = xs[idx];
      v4h h4 = { (_Float16)f.x, (_Float16)f.y, (_Float16)f.z, (_Float16)f.w };
      *(v4h*)(&As[row * K1_PAD + c4 * 4]) = h4;
    }
    __syncthreads();

#pragma unroll
    for (int mt = 0; mt < 4; ++mt) {
      v8f c = {0.f, 0.f, 0.f, 0.f, 0.f, 0.f, 0.f, 0.f};
      const _Float16* arow = &As[(mt * 16 + la) * K1_PAD];
#pragma unroll
      for (int ks = 0; ks < 8; ++ks) {
        v16h a = load_a_lds(arow, ks, hi16);
        c = __builtin_amdgcn_wmma_f32_16x16x32_f16(false, a, false, bf[ks],
                                                   (short)0, c, false, false);
      }
#pragma unroll
      for (int r = 0; r < 8; ++r) {     // D layout: lanes 0-15 -> M=r, lanes 16-31 -> M=r+8
        int m = mt * 16 + ((lane < 16) ? r : r + 8);
        xp[(size_t)(R + m) * HID + n] = c[r] + bsum;
      }
    }
    __syncthreads();   // protect LDS restage of next chunk
  }
}

// ====================== Kernel 2: serial scan h_t = tanh(xp_t + h @ W_hh^T) ============
// Batch rows are independent -> 4 workgroups, each owns 16 batch rows for all 2048 steps.
// 512 threads = 16 waves; wave w owns n-tile w with W_hh fragments resident in VGPRs for
// the whole scan. h double-buffered as f16 in LDS; one __syncthreads per timestep.
// Reads xp from `io` and overwrites the same slot with h_t (read-before-write per t).
#define K2_PAD 264

__global__ __launch_bounds__(512) void rnn_scan_wmma(
    const float* __restrict__ Whh, float* __restrict__ io)
{
  __shared__ __align__(16) _Float16 hb[2 * 16 * K2_PAD];
  const int tid  = threadIdx.x;
  const int lane = tid & 31;
  const int wv   = tid >> 5;       // n-tile 0..15
  const int hi16 = lane >> 4;
  const int la   = lane & 15;
  const int n    = wv * 16 + la;
  const int mb   = blockIdx.x;     // batch tile: rows mb*16 .. mb*16+15

  // W_hh B-fragments (row n), fp32 -> f16 once, live in VGPRs for all 2048 steps
  const float* wrow = Whh + (size_t)n * HID;
  v16h bf[8];
#pragma unroll
  for (int ks = 0; ks < 8; ++ks) bf[ks] = load_b_f32(wrow, ks, hi16);

  // h_{-1} = 0
  for (int i = tid; i < (2 * 16 * K2_PAD) / 2; i += 512) ((unsigned int*)hb)[i] = 0u;
  __syncthreads();

#pragma unroll 1
  for (int t = 0; t < SEQ; ++t) {
    const int cur = (t & 1) * (16 * K2_PAD);
    const int nxt = ((t & 1) ^ 1) * (16 * K2_PAD);
    const size_t base = ((size_t)t * BATCH + mb * 16) * HID;

    // C init = xp_t tile (fp32 from global)
    v8f c;
#pragma unroll
    for (int r = 0; r < 8; ++r) {
      int ml = (lane < 16) ? r : r + 8;
      c[r] = io[base + (size_t)ml * HID + n];
    }
    // hide next step's xp latency behind this step's WMMA chain
    if (t + 1 < SEQ) {
#pragma unroll
      for (int r = 0; r < 8; r += 2) {
        int ml = (lane < 16) ? r : r + 8;
        __builtin_prefetch(&io[base + (size_t)(BATCH * HID) + (size_t)ml * HID + n], 0, 0);
      }
    }

    // 16x16 tile GEMM over K=256: 8 x v_wmma_f32_16x16x32_f16
    const _Float16* arow = &hb[cur + la * K2_PAD];
#pragma unroll
    for (int ks = 0; ks < 8; ++ks) {
      v16h a = load_a_lds(arow, ks, hi16);
      c = __builtin_amdgcn_wmma_f32_16x16x32_f16(false, a, false, bf[ks],
                                                 (short)0, c, false, false);
    }

    // tanh, write h_t to global (fp32, in place over xp) and to next LDS buffer (f16)
#pragma unroll
    for (int r = 0; r < 8; ++r) {
      int ml = (lane < 16) ? r : r + 8;
      float th = fast_tanhf(c[r]);
      io[base + (size_t)ml * HID + n] = th;
      hb[nxt + ml * K2_PAD + n] = (_Float16)th;
    }
    __syncthreads();   // all h_t in LDS before step t+1 reads it
  }
}

// ======================================================================================
extern "C" void kernel_launch(void* const* d_in, const int* in_sizes, int n_in,
                              void* d_out, int out_size, void* d_ws, size_t ws_size,
                              hipStream_t stream) {
  (void)in_sizes; (void)n_in; (void)out_size; (void)d_ws; (void)ws_size;
  const float* x   = (const float*)d_in[0];   // [2048,64,256]
  const float* Wih = (const float*)d_in[1];   // [256,256]
  const float* Whh = (const float*)d_in[2];   // [256,256]
  const float* bih = (const float*)d_in[3];   // [256]
  const float* bhh = (const float*)d_in[4];   // [256]
  float* out = (float*)d_out;                 // [2048,64,256], holds xp then h

  // Phase 1: big parallel GEMM, xp -> d_out
  rnn_xproj_wmma<<<1024, 512, 0, stream>>>(x, Wih, bih, bhh, out);
  // Phase 2: 4 independent batch-tile scans, in place over d_out
  rnn_scan_wmma<<<4, 512, 0, stream>>>(Whh, out);
}
